// MultiHeadSelfAttention_26525718020546
// MI455X (gfx1250) — matrix operand order repaired and checked by reading, not compiled, not run on hardware
//
#include <hip/hip_runtime.h>

typedef __bf16 bf16;
typedef bf16 v16bf __attribute__((ext_vector_type(16)));
typedef bf16 v8bf  __attribute__((ext_vector_type(8)));
typedef float v8f  __attribute__((ext_vector_type(8)));

#define B_  4
#define S_  2048
#define H_  16
#define DM_ 1024
#define DK_ 64

// ---- WMMA fragment loaders (wave32 layouts per CDNA5 ISA 7.12.2) ----
__device__ __forceinline__ v16bf load_a_frag(const bf16* t, int stride, int row0, int k0) {
    int lane = threadIdx.x & 31;
    int hi   = lane >> 4;
    const bf16* p = t + (size_t)(row0 + (lane & 15)) * stride + k0 + hi * 8;
    v16bf r;
    *(v8bf*)&r         = *(const v8bf*)p;        // K = k0 + hi*8 .. +7
    *(((v8bf*)&r) + 1) = *(const v8bf*)(p + 16); // K = k0 + 16 + hi*8 .. +7
    return r;
}

__device__ __forceinline__ v16bf load_b_fragT(const bf16* t, int stride, int n0, int k0) {
    int lane = threadIdx.x & 31;
    int kh   = lane >> 4;
    const bf16* p = t + (size_t)(n0 + (lane & 15)) * stride + k0 + kh * 8;
    v16bf r;
    *(v8bf*)&r         = *(const v8bf*)p;
    *(((v8bf*)&r) + 1) = *(const v8bf*)(p + 16);
    return r;
}

// ---- gfx1250 async global->LDS copies (ASYNCcnt-tracked, bypass VGPRs) ----
__device__ __forceinline__ void async_copy_b128(void* lds, const void* g) {
    asm volatile("global_load_async_to_lds_b128 %0, %1, off"
                 :: "v"((unsigned)(size_t)lds), "v"(g) : "memory");
}
__device__ __forceinline__ void async_copy_b32(void* lds, const void* g) {
    asm volatile("global_load_async_to_lds_b32 %0, %1, off"
                 :: "v"((unsigned)(size_t)lds), "v"(g) : "memory");
}
__device__ __forceinline__ void wait_async0() {
    asm volatile("s_wait_asynccnt 0x0" ::: "memory");
}

// ---- kernel 0a: pack int32 mask into bits ----
__global__ __launch_bounds__(256) void pack_mask(const int* __restrict__ mask,
                                                 unsigned* __restrict__ out) {
    int idx = blockIdx.x * 256 + threadIdx.x;
    if (idx >= B_ * S_ * (S_ / 32)) return;
    const int* p = mask + (size_t)idx * 32;
    unsigned w = 0;
    #pragma unroll
    for (int i = 0; i < 32; ++i) w |= (p[i] != 0 ? 1u : 0u) << i;
    out[idx] = w;
}

// ---- kernel 0b: input fp32 -> bf16 ----
__global__ __launch_bounds__(256) void conv_x(const float* __restrict__ x, bf16* __restrict__ xb) {
    size_t i = ((size_t)blockIdx.x * 256 + threadIdx.x) * 4;   // 8M elems / 4
    float4 v = *(const float4*)(x + i);
    xb[i + 0] = (bf16)v.x; xb[i + 1] = (bf16)v.y;
    xb[i + 2] = (bf16)v.z; xb[i + 3] = (bf16)v.w;
}

// ---- kernel 0c: weights fp32 [h][d][n] -> bf16 transposed [m][h][n][d] ----
__global__ __launch_bounds__(256) void conv_w(const float* __restrict__ Wq,
                                              const float* __restrict__ Wk,
                                              const float* __restrict__ Wv,
                                              bf16* __restrict__ wb) {
    size_t i = (size_t)blockIdx.x * 256 + threadIdx.x;         // 3*16*1024*64
    int n = (int)(i & 63);
    int d = (int)((i >> 6) & 1023);
    int h = (int)((i >> 16) & 15);
    int m = (int)(i >> 20);
    const float* W = (m == 0) ? Wq : ((m == 1) ? Wk : Wv);
    wb[((size_t)(m * H_ + h) * DK_ + n) * DM_ + d] = (bf16)W[((size_t)h * DM_ + d) * DK_ + n];
}

// ---- kernel 1: QKV projection, async double-buffered, pure DMA + WMMA ----
#define QP_BUF(i)  ((i) * 24576)
#define QP_W_SUB   6144

__global__ __launch_bounds__(256) void qkv_proj(
    const bf16* __restrict__ xb,  const bf16* __restrict__ wb,
    const float* __restrict__ bq, const float* __restrict__ bk, const float* __restrict__ bv,
    bf16* __restrict__ qws, bf16* __restrict__ kws, bf16* __restrict__ vws)
{
    __shared__ alignas(16) char qpool[49152];

    const int s0 = blockIdx.x * 64, h = blockIdx.y, b = blockIdx.z;
    const int tid = threadIdx.x, wave = tid >> 5, lane = tid & 31;
    const int hi = lane >> 4, ln = lane & 15;

    const char* xg = (const char*)(xb + ((size_t)b * S_ + s0) * DM_);     // rows of 2048B
    const char* wg = (const char*)(wb + (size_t)h * DK_ * DM_);           // [m][h][n][d]
    const float* bia[3] = { bq + h * DK_, bk + h * DK_, bv + h * DK_ };

    auto issue_chunk = [&](int kc, int bi) {
        char* ib = qpool + QP_BUF(bi);
        {   // input tile: 64 rows x 32 k
            int r = tid >> 2, c = tid & 3;
            async_copy_b128(ib + r * 96 + c * 16, xg + (size_t)r * 2048 + kc * 64 + c * 16);
        }
        char* wbse = ib + QP_W_SUB;
        #pragma unroll
        for (int idx = tid; idx < 768; idx += 256) {  // 3 x (64 n x 4 chunks)
            int m = idx >> 8, n = (idx >> 2) & 63, c = idx & 3;
            async_copy_b128(wbse + m * 6144 + n * 96 + c * 16,
                            wg + ((size_t)m * H_ * DK_ + n) * 2048 + kc * 64 + c * 16);
        }
    };

    v8f acc[6] = {};   // 48 output tiles / 8 waves = 6 per wave

    issue_chunk(0, 0);
    wait_async0();
    __syncthreads();

    for (int kc = 0; kc < DM_ / 32; ++kc) {
        const int cur = kc & 1;
        if (kc + 1 < DM_ / 32) issue_chunk(kc + 1, 1 - cur);

        const bf16* it = (const bf16*)(qpool + QP_BUF(cur));
        const bf16* wt = (const bf16*)(qpool + QP_BUF(cur) + QP_W_SUB);

        // preload ALL fragments, then back-to-back WMMAs (avoids WMMA->WAR nops)
        v16bf af[6], bfr[6];
        #pragma unroll
        for (int i = 0; i < 6; ++i) {
            int t = wave + i * 8;
            int m = t >> 4, rem = t & 15, mi = rem >> 2, ni = rem & 3;
            af[i]  = load_a_frag(it, 48, mi * 16, 0);
            bfr[i] = load_b_fragT(wt + m * 3072, 48, ni * 16, 0);
        }
        #pragma unroll
        for (int i = 0; i < 6; ++i)
            acc[i] = __builtin_amdgcn_wmma_f32_16x16x32_bf16(false, af[i], false, bfr[i],
                                                             (short)0, acc[i], false, false);
        wait_async0();
        __syncthreads();
    }

    #pragma unroll
    for (int i = 0; i < 6; ++i) {
        int t = wave + i * 8;
        int m = t >> 4, rem = t & 15, mi = rem >> 2, ni = rem & 3;
        int n = ni * 16 + ln;
        float bval = bia[m][n];
        if (m < 2) {                               // Q,K: [b][h][s][64]
            bf16* op = (m == 0 ? qws : kws) + (((size_t)b * H_ + h) * S_ + s0 + mi * 16) * DK_;
            #pragma unroll
            for (int e = 0; e < 8; ++e)
                op[(size_t)(e + 8 * hi) * DK_ + n] = (bf16)(acc[i][e] + bval);
        } else {                                   // V: transposed [b][h][dv][S]
            bf16* op = vws + ((size_t)(b * H_ + h) * DK_ + n) * S_ + s0 + mi * 16;
            #pragma unroll
            for (int e = 0; e < 8; ++e)
                op[e + 8 * hi] = (bf16)(acc[i][e] + bval);
        }
    }
}

// ---- kernel 2: flash attention; async double-buffered K/V/mask tiles ----
#define PQ_OFF   0
#define BUF_OFF(i)  (20480 + (i) * 20480)
#define VT_SUB   10240
#define MK_OFF(i)   (61440 + (i) * 1024)

__global__ __launch_bounds__(256) void flash_attn(
    const bf16* __restrict__ qws, const bf16* __restrict__ kws, const bf16* __restrict__ vws,
    const unsigned* __restrict__ mws, float* __restrict__ out)
{
    __shared__ alignas(16) char pool[63488];

    const int q0 = blockIdx.x * 128, h = blockIdx.y, b = blockIdx.z;
    const int tid = threadIdx.x, wave = tid >> 5, lane = tid & 31;
    const int hi = lane >> 4, ln = lane & 15;
    const int r0 = wave * 16;

    const size_t headbase = ((size_t)b * H_ + h) * S_;
    const unsigned* mrow  = mws + ((size_t)b * S_ + q0) * (S_ / 32);
    const char* vghead    = (const char*)(vws + (size_t)(b * H_ + h) * DK_ * S_);

    auto issue_tiles = [&](int jj, int bi) {
        char* kt_b = pool + BUF_OFF(bi);
        char* vt_b = kt_b + VT_SUB;
        const char* kg = (const char*)(kws + (headbase + (size_t)jj * 64) * DK_);
        #pragma unroll
        for (int idx = tid; idx < 512; idx += 256) {            // K tile [t][d]
            int t = idx >> 3, c = idx & 7;
            async_copy_b128(kt_b + t * 160 + c * 16, kg + idx * 16);
        }
        const char* vg = vghead + (size_t)jj * 64 * 2;          // V^T tile [dv][t]
        #pragma unroll
        for (int idx = tid; idx < 512; idx += 256) {
            int dv = idx >> 3, c = idx & 7;
            async_copy_b128(vt_b + dv * 160 + c * 16, vg + (size_t)dv * S_ * 2 + c * 16);
        }
        {                                                       // mask: 128 rows x 2 words
            int r = tid >> 1, w = tid & 1;
            async_copy_b32(pool + MK_OFF(bi) + tid * 4,
                           mrow + (size_t)r * (S_ / 32) + jj * 2 + w);
        }
    };

    {   // prologue: Q tile (async) into P-staging region + first K/V/mask buffer
        const char* qg = (const char*)(qws + (headbase + q0) * DK_);
        #pragma unroll
        for (int idx = tid; idx < 1024; idx += 256) {           // 128 rows x 8 chunks
            int r = idx >> 3, c = idx & 7;
            async_copy_b128(pool + PQ_OFF + r * 160 + c * 16, qg + idx * 16);
        }
        issue_tiles(0, 0);
        wait_async0();
        __syncthreads();
    }

    const bf16* pq = (const bf16*)(pool + PQ_OFF);
    v16bf a0 = load_a_frag(pq, 80, r0, 0);                      // loop-invariant Q frags
    v16bf a1 = load_a_frag(pq, 80, r0, 32);

    float mrun[8], lrun[8];
    v8f O[4] = {};
    #pragma unroll
    for (int e = 0; e < 8; ++e) { mrun[e] = -3.0e38f; lrun[e] = 0.0f; }

    for (int j = 0; j < S_ / 64; ++j) {
        const int cur = j & 1;
        if (j + 1 < S_ / 64) issue_tiles(j + 1, 1 - cur);

        const bf16* kt = (const bf16*)(pool + BUF_OFF(cur));
        const bf16* vt = (const bf16*)(pool + BUF_OFF(cur) + VT_SUB);
        const unsigned* mk = (const unsigned*)(pool + MK_OFF(cur));

        // ---- S = (Q K^T)/8 : preload all 8 B-frags, then 8 independent WMMAs ----
        v16bf kb[8];
        #pragma unroll
        for (int ni = 0; ni < 4; ++ni) {
            kb[2 * ni]     = load_b_fragT(kt, 80, ni * 16, 0);
            kb[2 * ni + 1] = load_b_fragT(kt, 80, ni * 16, 32);
        }
        v8f sacc[4] = {};
        #pragma unroll
        for (int ni = 0; ni < 4; ++ni)
            sacc[ni] = __builtin_amdgcn_wmma_f32_16x16x32_bf16(false, a0, false, kb[2 * ni],
                                                               (short)0, sacc[ni], false, false);
        #pragma unroll
        for (int ni = 0; ni < 4; ++ni)
            sacc[ni] = __builtin_amdgcn_wmma_f32_16x16x32_bf16(false, a1, false, kb[2 * ni + 1],
                                                               (short)0, sacc[ni], false, false);

        // scale + mask (uint2 mask word per row)
        float sreg[4][8];
        #pragma unroll
        for (int e = 0; e < 8; ++e) {
            int r = r0 + e + 8 * hi;
            uint2 mw = *(const uint2*)&mk[r * 2];
            #pragma unroll
            for (int ni = 0; ni < 4; ++ni) {
                int c = ni * 16 + ln;
                unsigned word = (c < 32) ? mw.x : mw.y;
                float v = sacc[ni][e] * 0.125f;
                sreg[ni][e] = ((word >> (c & 31)) & 1u) ? v : -1.0e30f;
            }
        }

        // online softmax (columns of a row live across a 16-lane half)
        #pragma unroll
        for (int e = 0; e < 8; ++e) {
            float mx = fmaxf(fmaxf(sreg[0][e], sreg[1][e]), fmaxf(sreg[2][e], sreg[3][e]));
            #pragma unroll
            for (int d = 1; d < 16; d <<= 1) mx = fmaxf(mx, __shfl_xor(mx, d, 32));
            float mnew  = fmaxf(mrun[e], mx);
            float alpha = __expf(mrun[e] - mnew);
            float psum = 0.0f;
            #pragma unroll
            for (int ni = 0; ni < 4; ++ni) {
                float p = __expf(sreg[ni][e] - mnew);
                sreg[ni][e] = p;
                psum += p;
            }
            #pragma unroll
            for (int d = 1; d < 16; d <<= 1) psum += __shfl_xor(psum, d, 32);
            lrun[e] = lrun[e] * alpha + psum;
            mrun[e] = mnew;
            #pragma unroll
            for (int nv = 0; nv < 4; ++nv) O[nv][e] *= alpha;
        }

        // P: C-layout regs -> bf16 A-layout via per-wave LDS slice
        bf16* pw = (bf16*)(pool + PQ_OFF) + wave * 16 * 80;
        #pragma unroll
        for (int ni = 0; ni < 4; ++ni)
            #pragma unroll
            for (int e = 0; e < 8; ++e)
                pw[(e + 8 * hi) * 80 + ni * 16 + ln] = (bf16)sreg[ni][e];
        asm volatile("s_wait_dscnt 0x0" ::: "memory");
        v16bf pa0 = load_a_frag(pw, 80, 0, 0);
        v16bf pa1 = load_a_frag(pw, 80, 0, 32);

        // ---- O += P V : preload all 8 V B-frags, then 8 independent WMMAs ----
        v16bf vb[8];
        #pragma unroll
        for (int nv = 0; nv < 4; ++nv) {
            vb[2 * nv]     = load_b_fragT(vt, 80, nv * 16, 0);
            vb[2 * nv + 1] = load_b_fragT(vt, 80, nv * 16, 32);
        }
        #pragma unroll
        for (int nv = 0; nv < 4; ++nv)
            O[nv] = __builtin_amdgcn_wmma_f32_16x16x32_bf16(false, pa0, false, vb[2 * nv],
                                                            (short)0, O[nv], false, false);
        #pragma unroll
        for (int nv = 0; nv < 4; ++nv)
            O[nv] = __builtin_amdgcn_wmma_f32_16x16x32_bf16(false, pa1, false, vb[2 * nv + 1],
                                                            (short)0, O[nv], false, false);

        wait_async0();
        __syncthreads();
    }

    // epilogue: O / l ; fully-masked rows -> 0 (matches ref NaN->0)
    #pragma unroll
    for (int e = 0; e < 8; ++e) {
        float sc = (mrun[e] > -5.0e29f && lrun[e] > 0.0f) ? (1.0f / lrun[e]) : 0.0f;
        int r = q0 + r0 + e + 8 * hi;
        float* orow = out + ((size_t)b * S_ + r) * (H_ * DK_) + h * DK_;
        #pragma unroll
        for (int nv = 0; nv < 4; ++nv)
            orow[nv * 16 + ln] = O[nv][e] * sc;
    }
}

extern "C" void kernel_launch(void* const* d_in, const int* in_sizes, int n_in,
                              void* d_out, int out_size, void* d_ws, size_t ws_size,
                              hipStream_t stream) {
    (void)in_sizes; (void)n_in; (void)out_size; (void)ws_size;
    const float* x  = (const float*)d_in[0];
    const int* mask = (const int*)d_in[1];
    const float* Wq = (const float*)d_in[2];
    const float* bq = (const float*)d_in[3];
    const float* Wk = (const float*)d_in[4];
    const float* bk = (const float*)d_in[5];
    const float* Wv = (const float*)d_in[6];
    const float* bv = (const float*)d_in[7];
    float* out = (float*)d_out;

    char* ws = (char*)d_ws;
    bf16*     qws = (bf16*)(ws);
    bf16*     kws = (bf16*)(ws + (size_t)16 * 1024 * 1024);
    bf16*     vws = (bf16*)(ws + (size_t)32 * 1024 * 1024);
    unsigned* mws = (unsigned*)(ws + (size_t)48 * 1024 * 1024);
    bf16*     xbf = (bf16*)(ws + (size_t)50 * 1024 * 1024);
    bf16*     wbf = (bf16*)(ws + (size_t)66 * 1024 * 1024);

    pack_mask<<<dim3((B_ * S_ * (S_ / 32)) / 256), 256, 0, stream>>>(mask, mws);
    conv_x<<<dim3((B_ * S_ * DM_) / (4 * 256)), 256, 0, stream>>>(x, xbf);
    conv_w<<<dim3((3 * H_ * DM_ * DK_) / 256), 256, 0, stream>>>(Wq, Wk, Wv, wbf);
    qkv_proj<<<dim3(S_ / 64, H_, B_), 256, 0, stream>>>(xbf, wbf, bq, bk, bv, qws, kws, vws);
    flash_attn<<<dim3(S_ / 128, H_, B_), 256, 0, stream>>>(qws, kws, vws, mws, out);
}
